// DotProductAttention_64862596104700
// MI455X (gfx1250) — matrix operand order repaired and checked by reading, not compiled, not run on hardware
//
#include <hip/hip_runtime.h>

// CDNA5 / gfx1250 flash-attention: f16 WMMA + f32 accumulate,
// LDS double-buffered K/V staging, async-to-LDS mask staging (ASYNCcnt),
// softmax denominator fused into the PV GEMM via a ones-column WMMA,
// softmax in base-2 domain using the native v_exp_f32 (log2e pre-folded into Q).

typedef __attribute__((ext_vector_type(16))) _Float16 v16h;
typedef __attribute__((ext_vector_type(8)))  float    v8f;

#define B_   4
#define H_   16
#define S_   2048
#define D_   128
#define KT   32            // keys per tile
#define QW   16            // query rows per wave
#define WAVES 8
#define QB   (QW * WAVES)  // 128 query rows per block
#define NT   (S_ / KT)     // 64 key tiles
#define KPAD 136           // padded row stride (halves) for K tile   (272B, 16B aligned)
#define VTP  40            // padded row stride (halves) for V^T tile (80B, 16B aligned)
#define MPAD 36            // padded row stride (ints) for mask tile  (144B, 16B aligned)
#define PPAD 34            // padded row stride (halves) for per-wave P scratch

// Native hardware exp2 (v_exp_f32). Args are always <= 0; flush-to-zero for
// very negative inputs is exactly the masked-softmax semantics we want.
__device__ __forceinline__ float fast_exp2(float x) {
    return __builtin_amdgcn_exp2f(x);
}

__device__ __forceinline__ void issue_mask_async(const int* mrow0, int k0, int lane,
                                                 const int* smw)
{
    // Stage this wave's 16 mask rows (32 ints each) into its private LDS region.
    // 4 x global_load_async_to_lds_b128 per lane => 16 rows x 128B, coalesced.
    #pragma unroll
    for (int i = 0; i < 4; ++i) {
        int chunk = lane + 32 * i;         // 0..127 int4 chunks
        int rl    = chunk >> 3;            // row 0..15
        int c4    = (chunk & 7) << 2;      // col 0,4,...,28
        const int* g = mrow0 + (size_t)rl * S_ + k0 + c4;
        unsigned lds = (unsigned)(unsigned long long)(smw + rl * MPAD + c4);
        asm volatile("global_load_async_to_lds_b128 %0, %1, off"
                     :: "v"(lds), "v"(g) : "memory");
    }
}

__launch_bounds__(256)
__global__ void fa_wmma_kernel(const float* __restrict__ q,
                               const float* __restrict__ k,
                               const float* __restrict__ v,
                               const int*   __restrict__ mask,
                               float* __restrict__ out)
{
    __shared__ _Float16 sK [2][KT * KPAD];        // [buf][key][d]
    __shared__ _Float16 sVt[2][D_ * VTP];         // [buf][d][key]
    __shared__ int      sMask[QB * MPAD];         // [qrow_local][key_local]
    __shared__ _Float16 sP[WAVES * 16 * PPAD];    // per-wave P scratch

    const int tid  = threadIdx.x;
    const int wave = tid >> 5;
    const int lane = tid & 31;
    const int m    = lane & 15;
    const int hi   = lane >> 4;

    const int bh = blockIdx.y;          // b*H + h
    const int b  = bh >> 4;             // H_ == 16
    const int q0 = blockIdx.x * QB;

    const size_t base = (size_t)bh * S_ * D_;
    const float* qp = q + base;
    const float* kp = k + base;
    const float* vp = v + base;
    const int qrow0 = q0 + wave * QW;
    const int* mrow0 = mask + (size_t)b * S_ * S_ + (size_t)qrow0 * S_;
    const int* smw   = &sMask[wave * QW * MPAD];

    // 1/sqrt(128) * log2(e): softmax runs in the base-2 domain so the
    // hardware exp2 needs no per-exp multiply.
    const float qscale = 0.08838834764831845f * 1.4426950408889634f;

    // ---- Q fragments: A-layout, pre-scaled ----
    v16h aq[4];
    {
        const float* qr = qp + (size_t)(qrow0 + m) * D_;
        #pragma unroll
        for (int c = 0; c < 4; ++c) {
            #pragma unroll
            for (int j = 0; j < 16; ++j) {
                int d = 32 * c + ((j >= 8) ? 16 : 0) + hi * 8 + (j & 7);
                aq[c][j] = (_Float16)(qr[d] * qscale);
            }
        }
    }

    // constant all-ones B fragment (for the fused row-sum column block)
    v16h bones;
    #pragma unroll
    for (int j = 0; j < 16; ++j) bones[j] = (_Float16)1.0f;

    v8f oacc[9];   // [0..7]: O chunks; [8]: fused softmax denominator
    #pragma unroll
    for (int nc = 0; nc < 9; ++nc) oacc[nc] = (v8f){};
    float mrow[8];
    #pragma unroll
    for (int g = 0; g < 8; ++g) mrow[g] = -3.0e38f;

    // ---- prologue: async mask tile 0, stage K/V tile 0 into buffer 0 ----
    issue_mask_async(mrow0, 0, lane, smw);
    #pragma unroll
    for (int i = 0; i < 4; ++i) {
        int idx = tid + 256 * i;        // float4 index 0..1023
        int row = idx >> 5;             // key 0..31
        int col = (idx & 31) << 2;      // d 0..124
        float4 kf = *(const float4*)(kp + (size_t)row * D_ + col);
        float4 vf = *(const float4*)(vp + (size_t)row * D_ + col);
        _Float16* dk = &sK[0][row * KPAD + col];
        dk[0] = (_Float16)kf.x; dk[1] = (_Float16)kf.y;
        dk[2] = (_Float16)kf.z; dk[3] = (_Float16)kf.w;
        sVt[0][(col + 0) * VTP + row] = (_Float16)vf.x;
        sVt[0][(col + 1) * VTP + row] = (_Float16)vf.y;
        sVt[0][(col + 2) * VTP + row] = (_Float16)vf.z;
        sVt[0][(col + 3) * VTP + row] = (_Float16)vf.w;
    }
    __syncthreads();

    for (int kt = 0; kt < NT; ++kt) {
        const int k0  = kt * KT;
        const int cur = kt & 1;
        const int nxt = cur ^ 1;
        const bool more = (kt + 1 < NT);

        // ---- prefetch next K/V tile (global f32 -> registers) ----
        float4 kf[4], vf[4];
        if (more) {
            #pragma unroll
            for (int i = 0; i < 4; ++i) {
                int idx = tid + 256 * i;
                int row = idx >> 5;
                int col = (idx & 31) << 2;
                kf[i] = *(const float4*)(kp + (size_t)(k0 + KT + row) * D_ + col);
                vf[i] = *(const float4*)(vp + (size_t)(k0 + KT + row) * D_ + col);
            }
        }

        // ---- scores 16x32 = Qs(16x128) x K^T : 8 WMMAs ----
        const _Float16* skc = sK[cur];
        v8f sc0 = (v8f){}, sc1 = (v8f){};
        #pragma unroll
        for (int c = 0; c < 4; ++c) {
            v16h bk0, bk1;
            #pragma unroll
            for (int j = 0; j < 16; ++j) {
                int dl = 32 * c + ((j >= 8) ? 16 : 0) + hi * 8 + (j & 7);  // B-layout: K = (j>=8?16:0) + 8*hi + (j&7)
                bk0[j] = skc[(0  + m) * KPAD + dl];
                bk1[j] = skc[(16 + m) * KPAD + dl];
            }
            sc0 = __builtin_amdgcn_wmma_f32_16x16x32_f16(false, aq[c], false, bk0,
                                                         (short)0, sc0, false, false);
            sc1 = __builtin_amdgcn_wmma_f32_16x16x32_f16(false, aq[c], false, bk1,
                                                         (short)0, sc1, false, false);
        }

        // ---- mask (batched loads from this wave's async-staged LDS region) ----
        asm volatile("s_wait_asynccnt 0x0" ::: "memory");
        int mk[16];
        #pragma unroll
        for (int g = 0; g < 8; ++g) {
            mk[g]     = smw[(8 * hi + g) * MPAD + m];
            mk[8 + g] = smw[(8 * hi + g) * MPAD + 16 + m];
        }
        #pragma unroll
        for (int g = 0; g < 8; ++g) {
            sc0[g] = mk[g]     ? sc0[g] : -1.0e9f;
            sc1[g] = mk[8 + g] ? sc1[g] : -1.0e9f;
        }

        // ---- online softmax, base-2 domain (max only; denom fused in GEMM) ----
        float corr[8];
        #pragma unroll
        for (int g = 0; g < 8; ++g) {
            float tm = fmaxf(sc0[g], sc1[g]);
            #pragma unroll
            for (int off = 1; off < 16; off <<= 1)
                tm = fmaxf(tm, __shfl_xor(tm, off, 32));
            float mn = fmaxf(mrow[g], tm);
            corr[g] = fast_exp2(mrow[g] - mn);
            sc0[g] = fast_exp2(sc0[g] - mn);
            sc1[g] = fast_exp2(sc1[g] - mn);
            mrow[g] = mn;
        }
        #pragma unroll
        for (int nc = 0; nc < 9; ++nc)
            #pragma unroll
            for (int g = 0; g < 8; ++g)
                oacc[nc][g] *= corr[g];

        // ---- P: C-layout -> A-layout via per-wave LDS scratch ----
        _Float16* pw = &sP[wave * 16 * PPAD];
        #pragma unroll
        for (int g = 0; g < 8; ++g) {
            pw[(8 * hi + g) * PPAD + m]      = (_Float16)sc0[g];
            pw[(8 * hi + g) * PPAD + 16 + m] = (_Float16)sc1[g];
        }
        v16h pa;   // A 16x32: K = (j>=8?16:0) + 8*hi + (j&7)
        #pragma unroll
        for (int j = 0; j < 16; ++j) {
            int kl = ((j >= 8) ? 16 : 0) + hi * 8 + (j & 7);
            pa[j] = pw[m * PPAD + kl];
        }

        // ---- O(16x128) += P(16x32) x V(32x128) : 8 WMMAs,
        //      plus 1 WMMA accumulating the row sums (P x ones) ----
        const _Float16* svc = sVt[cur];
        #pragma unroll
        for (int nc = 0; nc < 8; ++nc) {
            v16h bv;
            #pragma unroll
            for (int j = 0; j < 16; ++j) {
                int kl = ((j >= 8) ? 16 : 0) + hi * 8 + (j & 7);  // B-layout K index
                bv[j] = svc[(nc * 16 + m) * VTP + kl];
            }
            oacc[nc] = __builtin_amdgcn_wmma_f32_16x16x32_f16(false, pa, false, bv,
                                                              (short)0, oacc[nc], false, false);
        }
        oacc[8] = __builtin_amdgcn_wmma_f32_16x16x32_f16(false, pa, false, bones,
                                                         (short)0, oacc[8], false, false);

        if (more) {
            // All of this wave's mask ds-reads are consumed above -> safe to
            // overwrite our private region with next tile's async loads.
            issue_mask_async(mrow0, k0 + KT, lane, smw);

            // cvt + store prefetched K/V into the other LDS buffer
            _Float16* skn = sK[nxt];
            _Float16* svn = sVt[nxt];
            #pragma unroll
            for (int i = 0; i < 4; ++i) {
                int idx = tid + 256 * i;
                int row = idx >> 5;
                int col = (idx & 31) << 2;
                _Float16* dk = &skn[row * KPAD + col];
                dk[0] = (_Float16)kf[i].x; dk[1] = (_Float16)kf[i].y;
                dk[2] = (_Float16)kf[i].z; dk[3] = (_Float16)kf[i].w;
                svn[(col + 0) * VTP + row] = (_Float16)vf[i].x;
                svn[(col + 1) * VTP + row] = (_Float16)vf[i].y;
                svn[(col + 2) * VTP + row] = (_Float16)vf[i].z;
                svn[(col + 3) * VTP + row] = (_Float16)vf[i].w;
            }
        }
        __syncthreads();
    }

    // ---- epilogue: normalize by the fused denominator and store ----
    // Every column of oacc[8] holds the row sum, so each lane already has it.
    float* op = out + base;
    #pragma unroll
    for (int g = 0; g < 8; ++g) {
        float inv = 1.0f / oacc[8][g];
        int row = qrow0 + 8 * hi + g;
        #pragma unroll
        for (int nc = 0; nc < 8; ++nc)
            op[(size_t)row * D_ + nc * 16 + m] = oacc[nc][g] * inv;
    }
}

extern "C" void kernel_launch(void* const* d_in, const int* in_sizes, int n_in,
                              void* d_out, int out_size, void* d_ws, size_t ws_size,
                              hipStream_t stream) {
    (void)in_sizes; (void)n_in; (void)out_size; (void)d_ws; (void)ws_size;
    const float* q    = (const float*)d_in[0];
    const float* k    = (const float*)d_in[1];
    const float* v    = (const float*)d_in[2];
    const int*   mask = (const int*)d_in[3];
    float* out = (float*)d_out;

    dim3 grid(S_ / QB, B_ * H_);   // (16, 64)
    dim3 block(256);
    fa_wmma_kernel<<<grid, block, 0, stream>>>(q, k, v, mask, out);
}
